// LinearTransformerEncoder_58600533786924
// MI455X (gfx1250) — matrix operand order repaired and checked by reading, not compiled, not run on hardware
//
#include <hip/hip_runtime.h>
#include <math.h>

// ---------------------------------------------------------------------------
// Linear Transformer Encoder for MI455X (gfx1250, wave32, WMMA).
// GEMMs use v_wmma_f32_16x16x32_bf16 with a bf16x3 (hi/lo) split for
// near-fp32 accuracy at the chip's bf16 matrix rate.
// Causal linear attention uses a chunked prefix scan (chunk=64) to expose
// 16x more parallelism than the naive sequential recurrence.
// ---------------------------------------------------------------------------

typedef __attribute__((ext_vector_type(16))) __bf16 v16bf;
typedef __attribute__((ext_vector_type(8)))  float  v8f;

#define TOKENS 2048   // B*S
#define DMODEL 512
#define KPRE   256    // F=240 padded to 256
#define NHEADS 16
#define DHEAD  32
#define CHUNK  64
#define NCHUNK 16     // 1024 / CHUNK

// ---- bf16 bit helpers (round-to-nearest-even), ushort storage -------------
__device__ __forceinline__ unsigned short f2bf_bits(float x) {
    union { float f; unsigned u; } a; a.f = x;
    unsigned r = a.u + 0x7FFFu + ((a.u >> 16) & 1u);
    return (unsigned short)(r >> 16);
}
__device__ __forceinline__ float bfbits2f(unsigned short s) {
    union { unsigned u; float f; } a; a.u = ((unsigned)s) << 16;
    return a.f;
}

// ---------------------------------------------------------------------------
// Weight conversion: W[K,N] fp32 row-major  ->  Wt_hi/Wt_lo[N,Kp] bf16,
// transposed + zero-padded in K so B-fragments load contiguously.
// ---------------------------------------------------------------------------
__global__ __launch_bounds__(256) void wconv_kernel(
    const float* __restrict__ W, int K, int N, int Kp,
    unsigned short* __restrict__ hi, unsigned short* __restrict__ lo) {
    int idx = blockIdx.x * 256 + threadIdx.x;
    if (idx >= N * Kp) return;
    int n = idx / Kp, k = idx - n * Kp;
    float v = (k < K) ? W[(size_t)k * N + n] : 0.f;
    unsigned short h = f2bf_bits(v);
    hi[idx] = h;
    lo[idx] = f2bf_bits(v - bfbits2f(h));
}

// Activation conversion: A[M,K] fp32 -> Ahi/Alo[M,K] bf16 (row-major).
__global__ __launch_bounds__(256) void aconv_kernel(
    const float* __restrict__ A, unsigned short* __restrict__ hi,
    unsigned short* __restrict__ lo, int n) {
    int idx = blockIdx.x * 256 + threadIdx.x;
    if (idx >= n) return;
    float v = A[idx];
    unsigned short h = f2bf_bits(v);
    hi[idx] = h;
    lo[idx] = f2bf_bits(v - bfbits2f(h));
}

// ---------------------------------------------------------------------------
// Fragment loaders. 16 bf16 elements per lane as two 16B chunks.
//   A (16x32, bf16): lane<16 holds K0..15, lane>=16 K16..31 (contiguous)
//     -> base offset 16*hiHalf, second chunk at +8.
//   B (32x16, bf16): lane<16 holds K0..15, lane>=16 K16..31 (contiguous)
//     -> base offset 16*hiHalf, second chunk at +8.
// ---------------------------------------------------------------------------
__device__ __forceinline__ v16bf load_frag(const unsigned short* p, int gap) {
    union { v16bf v; uint4 u[2]; } r;
    r.u[0] = *(const uint4*)(p);
    r.u[1] = *(const uint4*)(p + gap);
    return r.v;
}

// ---------------------------------------------------------------------------
// bf16x3 GEMM:  C[M,N] = A[M,K] * Wt^T + bias  (Wt stored [N,K]),
// optional ReLU. One wave -> 32x32 tile (2x2 WMMA tiles), 12 WMMA / K-step.
// ---------------------------------------------------------------------------
__global__ __launch_bounds__(256) void gemm_kernel(
    const unsigned short* __restrict__ Ahi, const unsigned short* __restrict__ Alo,
    int lda,
    const unsigned short* __restrict__ Bhi, const unsigned short* __restrict__ Blo,
    int ldb,
    const float* __restrict__ bias, float* __restrict__ C, int ldc,
    int M, int N, int K, int relu) {
    const int lane    = threadIdx.x & 31;
    const int wave    = threadIdx.x >> 5;
    const int l16     = lane & 15;
    const int hiHalf  = lane >> 4;           // 0 or 1 (per-lane load pattern)
    const int tilesM  = M >> 5;
    const int tid     = blockIdx.x * 8 + wave;
    if (tid >= tilesM * (N >> 5)) return;    // wave-uniform
    const int m0 = (tid % tilesM) << 5;
    const int n0 = (tid / tilesM) << 5;

    const int koffA = hiHalf ? 16 : 0;
    const int koffB = hiHalf ? 16 : 0;

    const unsigned short* A0h = Ahi + (size_t)(m0 + l16) * lda + koffA;
    const unsigned short* A1h = A0h + (size_t)16 * lda;
    const unsigned short* A0l = Alo + (size_t)(m0 + l16) * lda + koffA;
    const unsigned short* A1l = A0l + (size_t)16 * lda;
    const unsigned short* B0h = Bhi + (size_t)(n0 + l16) * ldb + koffB;
    const unsigned short* B1h = B0h + (size_t)16 * ldb;
    const unsigned short* B0l = Blo + (size_t)(n0 + l16) * ldb + koffB;
    const unsigned short* B1l = B0l + (size_t)16 * ldb;

    v8f acc00 = {}; v8f acc01 = {}; v8f acc10 = {}; v8f acc11 = {};

    for (int k0 = 0; k0 < K; k0 += 32) {
        __builtin_prefetch(A0h + k0 + 64, 0, 1);
        __builtin_prefetch(B0h + k0 + 64, 0, 1);
        v16bf ah0 = load_frag(A0h + k0, 8);
        v16bf ah1 = load_frag(A1h + k0, 8);
        v16bf al0 = load_frag(A0l + k0, 8);
        v16bf al1 = load_frag(A1l + k0, 8);
        v16bf bh0 = load_frag(B0h + k0, 8);
        v16bf bh1 = load_frag(B1h + k0, 8);
        v16bf bl0 = load_frag(B0l + k0, 8);
        v16bf bl1 = load_frag(B1l + k0, 8);

        acc00 = __builtin_amdgcn_wmma_f32_16x16x32_bf16(false, ah0, false, bh0, (short)0, acc00, false, false);
        acc00 = __builtin_amdgcn_wmma_f32_16x16x32_bf16(false, ah0, false, bl0, (short)0, acc00, false, false);
        acc00 = __builtin_amdgcn_wmma_f32_16x16x32_bf16(false, al0, false, bh0, (short)0, acc00, false, false);

        acc01 = __builtin_amdgcn_wmma_f32_16x16x32_bf16(false, ah0, false, bh1, (short)0, acc01, false, false);
        acc01 = __builtin_amdgcn_wmma_f32_16x16x32_bf16(false, ah0, false, bl1, (short)0, acc01, false, false);
        acc01 = __builtin_amdgcn_wmma_f32_16x16x32_bf16(false, al0, false, bh1, (short)0, acc01, false, false);

        acc10 = __builtin_amdgcn_wmma_f32_16x16x32_bf16(false, ah1, false, bh0, (short)0, acc10, false, false);
        acc10 = __builtin_amdgcn_wmma_f32_16x16x32_bf16(false, ah1, false, bl0, (short)0, acc10, false, false);
        acc10 = __builtin_amdgcn_wmma_f32_16x16x32_bf16(false, al1, false, bh0, (short)0, acc10, false, false);

        acc11 = __builtin_amdgcn_wmma_f32_16x16x32_bf16(false, ah1, false, bh1, (short)0, acc11, false, false);
        acc11 = __builtin_amdgcn_wmma_f32_16x16x32_bf16(false, ah1, false, bl1, (short)0, acc11, false, false);
        acc11 = __builtin_amdgcn_wmma_f32_16x16x32_bf16(false, al1, false, bh1, (short)0, acc11, false, false);
    }

    // C/D layout: lane<16 -> row m0+r, col n0+lane ; lane>=16 -> row m0+8+r.
    const float b0 = bias ? bias[n0 + l16]      : 0.f;
    const float b1 = bias ? bias[n0 + 16 + l16] : 0.f;
    const int row0 = m0 + (hiHalf ? 8 : 0);
    const int col0 = n0 + l16;
#pragma unroll
    for (int r = 0; r < 8; ++r) {
        float v;
        v = acc00[r] + b0; if (relu) v = fmaxf(v, 0.f);
        C[(size_t)(row0 + r) * ldc + col0] = v;
        v = acc01[r] + b1; if (relu) v = fmaxf(v, 0.f);
        C[(size_t)(row0 + r) * ldc + col0 + 16] = v;
        v = acc10[r] + b0; if (relu) v = fmaxf(v, 0.f);
        C[(size_t)(row0 + 16 + r) * ldc + col0] = v;
        v = acc11[r] + b1; if (relu) v = fmaxf(v, 0.f);
        C[(size_t)(row0 + 16 + r) * ldc + col0 + 16] = v;
    }
}

// ---------------------------------------------------------------------------
// Input LayerNorm over F=240 features, zero-padded to 256 columns.
// ---------------------------------------------------------------------------
__global__ __launch_bounds__(256) void ln_in_kernel(
    const float* __restrict__ x, const float* __restrict__ g,
    const float* __restrict__ b, float* __restrict__ out) {
    __shared__ float red[8];
    const int t = blockIdx.x, tid = threadIdx.x;
    float v = (tid < 240) ? x[(size_t)t * 240 + tid] : 0.f;
    float s = v;
    for (int m = 16; m; m >>= 1) s += __shfl_xor(s, m, 32);
    if ((tid & 31) == 0) red[tid >> 5] = s;
    __syncthreads();
    float mean = (red[0]+red[1]+red[2]+red[3]+red[4]+red[5]+red[6]+red[7]) * (1.f/240.f);
    float d = (tid < 240) ? (v - mean) : 0.f;
    float s2 = d * d;
    __syncthreads();
    for (int m = 16; m; m >>= 1) s2 += __shfl_xor(s2, m, 32);
    if ((tid & 31) == 0) red[tid >> 5] = s2;
    __syncthreads();
    float var = (red[0]+red[1]+red[2]+red[3]+red[4]+red[5]+red[6]+red[7]) * (1.f/240.f);
    float y = 0.f;
    if (tid < 240) y = d * rsqrtf(var + 1e-5f) * g[tid] + b[tid];
    out[(size_t)t * KPRE + tid] = y;
}

// ---------------------------------------------------------------------------
// Fused residual-add + LayerNorm over D=512. a may be null (plain LN).
// ---------------------------------------------------------------------------
__global__ __launch_bounds__(256) void add_ln_kernel(
    float* __restrict__ h, const float* __restrict__ a,
    const float* __restrict__ g, const float* __restrict__ b) {
    __shared__ float red[8];
    const int t = blockIdx.x, tid = threadIdx.x;
    const size_t base = (size_t)t * DMODEL;
    float v0 = h[base + tid]       + (a ? a[base + tid]       : 0.f);
    float v1 = h[base + tid + 256] + (a ? a[base + tid + 256] : 0.f);
    float s = v0 + v1;
    for (int m = 16; m; m >>= 1) s += __shfl_xor(s, m, 32);
    if ((tid & 31) == 0) red[tid >> 5] = s;
    __syncthreads();
    float mean = (red[0]+red[1]+red[2]+red[3]+red[4]+red[5]+red[6]+red[7]) * (1.f/512.f);
    float d0 = v0 - mean, d1 = v1 - mean;
    float s2 = d0*d0 + d1*d1;
    __syncthreads();
    for (int m = 16; m; m >>= 1) s2 += __shfl_xor(s2, m, 32);
    if ((tid & 31) == 0) red[tid >> 5] = s2;
    __syncthreads();
    float var = (red[0]+red[1]+red[2]+red[3]+red[4]+red[5]+red[6]+red[7]) * (1.f/512.f);
    float r = rsqrtf(var + 1e-5f);
    h[base + tid]       = d0 * r * g[tid]       + b[tid];
    h[base + tid + 256] = d1 * r * g[tid + 256] + b[tid + 256];
}

// ---------------------------------------------------------------------------
// Chunked causal linear attention (fast_transformers): phi(x)=elu(x)+1.
// State per (b,head): KV[32,32] = sum K_s (x) V_s, and ksum[32] = sum K_s.
// Chunk scan: per-chunk sums -> exclusive prefix over chunks -> in-chunk
// sequential recurrence seeded by the prefix. Exact fp32.
// Thread layout (phase 1/3): lane = feature dim d; wave w owns V-dims 4w..4w+3.
// ---------------------------------------------------------------------------
__device__ __forceinline__ float phi_map(float x) {
    return (x > 0.f) ? (x + 1.f) : __expf(x);
}

// Phase 1: per-chunk sums. grid = 32 bh * 16 chunks.
__global__ __launch_bounds__(256) void attn_chunk_sums(
    const float* __restrict__ k, const float* __restrict__ v,
    float* __restrict__ chunkKV, float* __restrict__ chunkKsum) {
    const int lane = threadIdx.x & 31;   // d
    const int mg   = threadIdx.x >> 5;   // value-dim group
    const int bh   = blockIdx.x >> 4;
    const int c    = blockIdx.x & 15;
    const int bb   = bh >> 4, hd = bh & 15;
    const size_t base = (size_t)bb * 1024 * DMODEL + (size_t)hd * DHEAD
                      + (size_t)c * CHUNK * DMODEL;
    float kv0 = 0.f, kv1 = 0.f, kv2 = 0.f, kv3 = 0.f, ksum = 0.f;
    for (int s = 0; s < CHUNK; ++s) {
        const size_t off = base + (size_t)s * DMODEL;
        float Kf = phi_map(k[off + lane]);
        const float4 vv = *(const float4*)(v + off + mg * 4);
        ksum += Kf;
        kv0 = fmaf(Kf, vv.x, kv0);
        kv1 = fmaf(Kf, vv.y, kv1);
        kv2 = fmaf(Kf, vv.z, kv2);
        kv3 = fmaf(Kf, vv.w, kv3);
    }
    const size_t cb = ((size_t)bh * NCHUNK + c) * 32;   // 32 rows (d)
    *(float4*)(chunkKV + (cb + lane) * 32 + mg * 4) = make_float4(kv0, kv1, kv2, kv3);
    if (mg == 0) chunkKsum[cb + lane] = ksum;
}

// Phase 2: exclusive prefix over the 16 chunks. grid = 32 bh blocks.
__global__ __launch_bounds__(256) void attn_prefix(
    const float* __restrict__ chunkKV, const float* __restrict__ chunkKsum,
    float* __restrict__ preKV, float* __restrict__ preKsum) {
    const int bh = blockIdx.x, tid = threadIdx.x;
    const size_t stride = 32 * 32;                       // floats per chunk state
    const size_t basekv = (size_t)bh * NCHUNK * stride + (size_t)tid * 4;
    float4 run = make_float4(0.f, 0.f, 0.f, 0.f);
    for (int c = 0; c < NCHUNK; ++c) {
        const size_t o = basekv + (size_t)c * stride;
        *(float4*)(preKV + o) = run;
        const float4 t = *(const float4*)(chunkKV + o);
        run.x += t.x; run.y += t.y; run.z += t.z; run.w += t.w;
    }
    if (tid < 32) {
        float rk = 0.f;
        const size_t baseks = (size_t)bh * NCHUNK * 32 + tid;
        for (int c = 0; c < NCHUNK; ++c) {
            preKsum[baseks + (size_t)c * 32] = rk;
            rk += chunkKsum[baseks + (size_t)c * 32];
        }
    }
}

// Phase 3: in-chunk recurrence seeded by prefix state. grid = 32 bh * 16 chunks.
__global__ __launch_bounds__(256) void attn_chunk_scan(
    const float* __restrict__ q, const float* __restrict__ k,
    const float* __restrict__ v, const float* __restrict__ preKV,
    const float* __restrict__ preKsum, float* __restrict__ o) {
    const int lane = threadIdx.x & 31;   // d
    const int mg   = threadIdx.x >> 5;
    const int bh   = blockIdx.x >> 4;
    const int c    = blockIdx.x & 15;
    const int bb   = bh >> 4, hd = bh & 15;
    const size_t base = (size_t)bb * 1024 * DMODEL + (size_t)hd * DHEAD
                      + (size_t)c * CHUNK * DMODEL;
    const size_t cb = ((size_t)bh * NCHUNK + c) * 32;
    const float4 kvi = *(const float4*)(preKV + (cb + lane) * 32 + mg * 4);
    float kv0 = kvi.x, kv1 = kvi.y, kv2 = kvi.z, kv3 = kvi.w;
    float ksum = preKsum[cb + lane];

    for (int s = 0; s < CHUNK; ++s) {
        const size_t off = base + (size_t)s * DMODEL;
        float Q  = phi_map(q[off + lane]);
        float Kf = phi_map(k[off + lane]);
        const float4 vv = *(const float4*)(v + off + mg * 4);
        ksum += Kf;
        kv0 = fmaf(Kf, vv.x, kv0);
        kv1 = fmaf(Kf, vv.y, kv1);
        kv2 = fmaf(Kf, vv.z, kv2);
        kv3 = fmaf(Kf, vv.w, kv3);
        float p0 = Q * kv0, p1 = Q * kv1, p2 = Q * kv2, p3 = Q * kv3, pz = Q * ksum;
        for (int m = 16; m; m >>= 1) {
            p0 += __shfl_xor(p0, m, 32);
            p1 += __shfl_xor(p1, m, 32);
            p2 += __shfl_xor(p2, m, 32);
            p3 += __shfl_xor(p3, m, 32);
            pz += __shfl_xor(pz, m, 32);
        }
        if (lane == 0) {
            float Z = 1.f / (pz + 1e-6f);
            float4 out4 = make_float4(p0 * Z, p1 * Z, p2 * Z, p3 * Z);
            *(float4*)(o + off + mg * 4) = out4;
        }
    }
}

// ---------------------------------------------------------------------------
// Host-side orchestration (graph-capture safe: all work on `stream`).
// ---------------------------------------------------------------------------
extern "C" void kernel_launch(void* const* d_in, const int* in_sizes, int n_in,
                              void* d_out, int out_size, void* d_ws, size_t ws_size,
                              hipStream_t stream) {
    const float* x      = (const float*)d_in[0];
    const float* g_in   = (const float*)d_in[1];
    const float* b_in   = (const float*)d_in[2];
    const float* W_pre  = (const float*)d_in[3];
    const float* b_pre  = (const float*)d_in[4];
    const float* Wq     = (const float*)d_in[5];
    const float* bq     = (const float*)d_in[6];
    const float* Wk     = (const float*)d_in[7];
    const float* bk     = (const float*)d_in[8];
    const float* Wv     = (const float*)d_in[9];
    const float* bv     = (const float*)d_in[10];
    const float* Wo     = (const float*)d_in[11];
    const float* bo     = (const float*)d_in[12];
    const float* W1     = (const float*)d_in[13];
    const float* bf1    = (const float*)d_in[14];
    const float* W2     = (const float*)d_in[15];
    const float* bf2    = (const float*)d_in[16];
    const float* ln1_g  = (const float*)d_in[17];
    const float* ln1_b  = (const float*)d_in[18];
    const float* ln2_g  = (const float*)d_in[19];
    const float* ln2_b  = (const float*)d_in[20];
    const float* lnf_g  = (const float*)d_in[21];
    const float* lnf_b  = (const float*)d_in[22];
    const float* W_post = (const float*)d_in[23];
    const float* b_post = (const float*)d_in[24];

    const size_t DD = (size_t)DMODEL * DMODEL;

    // ---- bump allocator over d_ws (~50 MB total) ----
    char* wp = (char*)d_ws;
    auto carve = [&](size_t bytes) -> void* {
        void* p = (void*)wp;
        wp += (bytes + 255) & ~(size_t)255;
        return p;
    };
    float* xln = (float*)carve((size_t)TOKENS * KPRE   * 4);
    float* h   = (float*)carve((size_t)TOKENS * DMODEL * 4);
    float* qb  = (float*)carve((size_t)TOKENS * DMODEL * 4);
    float* kb  = (float*)carve((size_t)TOKENS * DMODEL * 4);
    float* vb  = (float*)carve((size_t)TOKENS * DMODEL * 4);
    float* ab  = (float*)carve((size_t)TOKENS * DMODEL * 4);
    float* t1  = (float*)carve((size_t)TOKENS * DMODEL * 4);
    float* t2  = (float*)carve((size_t)TOKENS * DMODEL * 4);
    float* cKV  = (float*)carve((size_t)32 * NCHUNK * 32 * 32 * 4);
    float* pKV  = (float*)carve((size_t)32 * NCHUNK * 32 * 32 * 4);
    float* cKs  = (float*)carve((size_t)32 * NCHUNK * 32 * 4);
    float* pKs  = (float*)carve((size_t)32 * NCHUNK * 32 * 4);
    unsigned short* ahi = (unsigned short*)carve((size_t)TOKENS * DMODEL * 2);
    unsigned short* alo = (unsigned short*)carve((size_t)TOKENS * DMODEL * 2);
    unsigned short* pre_hi  = (unsigned short*)carve((size_t)DMODEL * KPRE * 2);
    unsigned short* pre_lo  = (unsigned short*)carve((size_t)DMODEL * KPRE * 2);
    unsigned short* post_hi = (unsigned short*)carve(DD * 2);
    unsigned short* post_lo = (unsigned short*)carve(DD * 2);
    unsigned short* whi[6];
    unsigned short* wlo[6];
    for (int j = 0; j < 6; ++j) {
        whi[j] = (unsigned short*)carve(DD * 2);
        wlo[j] = (unsigned short*)carve(DD * 2);
    }

    auto wconv = [&](const float* W, int K, int N, int Kp,
                     unsigned short* hi, unsigned short* lo) {
        int tot = N * Kp;
        wconv_kernel<<<(tot + 255) / 256, 256, 0, stream>>>(W, K, N, Kp, hi, lo);
    };
    // A has row-stride == K (256 for xln, 512 otherwise); C row-stride 512.
    auto gemm = [&](const float* Af, int K, const unsigned short* bhi,
                    const unsigned short* blo, const float* bias,
                    float* C, int relu) {
        int tot = TOKENS * K;
        aconv_kernel<<<(tot + 255) / 256, 256, 0, stream>>>(Af, ahi, alo, tot);
        gemm_kernel<<<128, 256, 0, stream>>>(ahi, alo, K, bhi, blo, K,
                                             bias, C, DMODEL, TOKENS, DMODEL, K, relu);
    };
    auto attn = [&]() {
        attn_chunk_sums<<<32 * NCHUNK, 256, 0, stream>>>(kb, vb, cKV, cKs);
        attn_prefix<<<32, 256, 0, stream>>>(cKV, cKs, pKV, pKs);
        attn_chunk_scan<<<32 * NCHUNK, 256, 0, stream>>>(qb, kb, vb, pKV, pKs, ab);
    };

    // ---- input LN + pre-projection (K padded 240 -> 256) ----
    wconv(W_pre, 240, DMODEL, KPRE, pre_hi, pre_lo);
    ln_in_kernel<<<TOKENS, 256, 0, stream>>>(x, g_in, b_in, xln);
    gemm(xln, KPRE, pre_hi, pre_lo, b_pre, h, 0);

    // ---- 6 encoder layers ----
    for (int l = 0; l < 6; ++l) {
        wconv(Wq + (size_t)l * DD, DMODEL, DMODEL, DMODEL, whi[0], wlo[0]);
        wconv(Wk + (size_t)l * DD, DMODEL, DMODEL, DMODEL, whi[1], wlo[1]);
        wconv(Wv + (size_t)l * DD, DMODEL, DMODEL, DMODEL, whi[2], wlo[2]);
        wconv(Wo + (size_t)l * DD, DMODEL, DMODEL, DMODEL, whi[3], wlo[3]);
        wconv(W1 + (size_t)l * DD, DMODEL, DMODEL, DMODEL, whi[4], wlo[4]);
        wconv(W2 + (size_t)l * DD, DMODEL, DMODEL, DMODEL, whi[5], wlo[5]);

        gemm(h, DMODEL, whi[0], wlo[0], bq + l * DMODEL, qb, 0);
        gemm(h, DMODEL, whi[1], wlo[1], bk + l * DMODEL, kb, 0);
        gemm(h, DMODEL, whi[2], wlo[2], bv + l * DMODEL, vb, 0);

        attn();

        gemm(ab, DMODEL, whi[3], wlo[3], bo + l * DMODEL, t1, 0);
        add_ln_kernel<<<TOKENS, 256, 0, stream>>>(h, t1, ln1_g + l * DMODEL, ln1_b + l * DMODEL);

        gemm(h, DMODEL, whi[4], wlo[4], bf1 + l * DMODEL, t2, 1);   // ReLU
        gemm(t2, DMODEL, whi[5], wlo[5], bf2 + l * DMODEL, t1, 0);
        add_ln_kernel<<<TOKENS, 256, 0, stream>>>(h, t1, ln2_g + l * DMODEL, ln2_b + l * DMODEL);
    }

    // ---- final LN + post projection into d_out ----
    wconv(W_post, DMODEL, DMODEL, DMODEL, post_hi, post_lo);
    add_ln_kernel<<<TOKENS, 256, 0, stream>>>(h, nullptr, lnf_g, lnf_b);
    gemm(h, DMODEL, post_hi, post_lo, b_post, (float*)d_out, 0);
}
